// FC_STGNN_Pedestrians_FC_75230647157506
// MI455X (gfx1250) — matrix operand order repaired and checked by reading, not compiled, not run on hardware
//
#include <hip/hip_runtime.h>

#define BS 4
#define NN 512
#define IN_DIM 256
#define H1 128
#define EMB 64

typedef _Float16 v16h __attribute__((ext_vector_type(16)));
typedef _Float16 h8   __attribute__((ext_vector_type(8)));
typedef float    v8f  __attribute__((ext_vector_type(8)));

// ---------------------------------------------------------------------------
// Kernel 1: node MLP (tiny).  h = leaky(x@W1 + b1); emb = leaky(h@W2 + b2)
// ---------------------------------------------------------------------------
__global__ void __launch_bounds__(256) node_mlp_kernel(
    const float* __restrict__ x,  const float* __restrict__ w1, const float* __restrict__ b1,
    const float* __restrict__ w2, const float* __restrict__ b2,
    float* __restrict__ embF32, _Float16* __restrict__ embF16)
{
    __shared__ float xs[IN_DIM];
    __shared__ float hs[H1];
    const int row = blockIdx.x;          // b*N + n  (2048 rows)
    const int t = threadIdx.x;

    xs[t] = x[(size_t)row * IN_DIM + t];
    __syncthreads();

    if (t < H1) {
        float s = b1[t];
        #pragma unroll 8
        for (int f = 0; f < IN_DIM; ++f) s = fmaf(xs[f], w1[f * H1 + t], s);
        hs[t] = s > 0.f ? s : 0.01f * s;
    }
    __syncthreads();

    if (t < EMB) {
        float s = b2[t];
        #pragma unroll 8
        for (int f = 0; f < H1; ++f) s = fmaf(hs[f], w2[f * EMB + t], s);
        s = s > 0.f ? s : 0.01f * s;
        embF32[(size_t)row * EMB + t] = s;
        embF16[(size_t)row * EMB + t] = (_Float16)s;
    }
}

// ---------------------------------------------------------------------------
// Edge-MLP helpers
// ---------------------------------------------------------------------------

// One 16-edge x 64-channel tile: 4 k-steps x 4 n-columns of
// v_wmma_f32_16x16x32_f16, A built on the fly from packed-f16 ops.
__device__ __forceinline__ void wmma_tile(v8f* acc, const h8* hiC, const h8* hjC,
                                          const v16h (&bf)[4][4])
{
    #pragma unroll
    for (int s = 0; s < 4; ++s) {
        // K block 32s..32s+31: s<2 -> (hi-hj) dims, s>=2 -> (hi*hj) dims
        const int m0 = (s & 1) * 2;
        union { v16h v; h8 h[2]; } a;
        if (s < 2) { a.h[0] = hiC[m0] - hjC[m0]; a.h[1] = hiC[m0 + 1] - hjC[m0 + 1]; }
        else       { a.h[0] = hiC[m0] * hjC[m0]; a.h[1] = hiC[m0 + 1] * hjC[m0 + 1]; }
        #pragma unroll
        for (int t = 0; t < 4; ++t)
            acc[t] = __builtin_amdgcn_wmma_f32_16x16x32_f16(
                false, a.v, false, bf[s][t], (short)0, acc[t], false, false);
    }
}

// Epilogue: eh = relu(acc + e1b); logits = eh @ e2w + e2b; softmax(2); scatter.
// Index-folding reduction: 16 partials/lane -> 15 shuffles; value k ends on
// lane nl==k  (edge r = nl>>1, class c = nl&1).
__device__ __forceinline__ void edge_epilogue(
    const v8f* acc, const float* bias, const float* w20, const float* w21,
    float eb0, float eb1, int kh, int nl, int b, int i, int jgbase, int jbLocal,
    float* __restrict__ adjRow, float* __restrict__ pred)
{
    float cur[16];                        // k = (r<<1) | c
    #pragma unroll
    for (int r = 0; r < 8; ++r) {
        float p0 = 0.f, p1 = 0.f;
        #pragma unroll
        for (int t = 0; t < 4; ++t) {
            float e = fmaxf(acc[t][r] + bias[t], 0.f);
            p0 = fmaf(e, w20[t], p0);
            p1 = fmaf(e, w21[t], p1);
        }
        cur[2 * r]     = p0;
        cur[2 * r + 1] = p1;
    }
    // fold bit s of k onto lane bit s (stays inside the 16-lane half)
    #pragma unroll
    for (int s = 0; s < 4; ++s) {
        const int  mask = 1 << s;
        const bool hi   = (nl >> s) & 1;
        #pragma unroll
        for (int j = 0; j < (8 >> s); ++j) {
            float keep = hi ? cur[2 * j + 1] : cur[2 * j];
            float send = hi ? cur[2 * j]     : cur[2 * j + 1];
            cur[j] = keep + __shfl_xor(send, mask, 32);
        }
    }
    const int c = nl & 1;
    const int r = nl >> 1;
    float logit = cur[0] + (c ? eb1 : eb0);
    float other = __shfl_xor(logit, 1, 32);
    float l0 = c ? other : logit;
    float l1 = c ? logit : other;
    float mx  = fmaxf(l0, l1);
    float x0  = __expf(l0 - mx), x1 = __expf(l1 - mx);
    float inv = 1.f / (x0 + x1);
    float q0  = x0 * inv, q1 = x1 * inv;

    if (c == 0) {                        // 16 writer lanes, coalesced over j
        const int M = kh * 8 + r;
        const int j = jgbase + jbLocal + M;
        adjRow[jbLocal + M] = q1;
        if (j != i) {
            int jp = j - (j > i);
            size_t pidx = (((size_t)b * NN * (NN - 1)) + (size_t)i * (NN - 1) + jp) * 2;
            pred[pidx]     = q0;
            pred[pidx + 1] = q1;
        }
    }
}

// ---------------------------------------------------------------------------
// Kernel 2: edge MLP over all (i,j) pairs via v_wmma_f32_16x16x32_f16.
// Two tiles in flight per iteration: tile-A epilogue VALU co-executes with
// tile-B's WMMAs (XDL/TRANS co-execution), hiding the WMMA->VALU hazard NOPs.
// ---------------------------------------------------------------------------
__global__ void __launch_bounds__(256) edge_mlp_kernel(
    const _Float16* __restrict__ embF16,
    const float* __restrict__ e1w, const float* __restrict__ e1b,
    const float* __restrict__ e2w, const float* __restrict__ e2b,
    float* __restrict__ adj, float* __restrict__ pred)
{
    __shared__ __align__(16) _Float16 sEmb[256 * EMB];     // 32 KB: emb slice for 256 j's
    __shared__ __align__(32) _Float16 sB[4 * 4 * 32 * 16]; // 16 KB: B frags [s][t][lane][16]

    const int bid = blockIdx.x;          // BS * 64 * 2 = 512 blocks
    const int b   = bid >> 7;
    const int rem = bid & 127;
    const int ig  = rem >> 1;            // i-group of 8 rows
    const int jg  = rem & 1;             // j half: 256 columns
    const int tid = threadIdx.x;

    // Stage the j-slice of emb (f16) into LDS
    const _Float16* esrc = embF16 + ((size_t)b * NN + jg * 256) * EMB;
    for (int idx = tid; idx < 256 * EMB / 8; idx += 256)
        ((h8*)sEmb)[idx] = ((const h8*)esrc)[idx];

    // Swizzle e1 weights into per-lane WMMA B fragments (32x16 f16):
    // lane l holds N = 16t + (l&15), K = 32s + 8*(l>>4) + {0..7, 16..23}
    for (int idx = tid; idx < 4 * 4 * 32; idx += 256) {
        int s  = idx >> 7;
        int tt = (idx >> 5) & 3;
        int ln = idx & 31;
        int Kb  = 32 * s + 8 * (ln >> 4);
        int col = 16 * tt + (ln & 15);
        _Float16* dst = &sB[idx * 16];
        #pragma unroll
        for (int e = 0; e < 16; ++e)
            dst[e] = (_Float16)e1w[(Kb + (e & 7) + 16 * (e >> 3)) * EMB + col];
    }
    __syncthreads();

    const int wave = tid >> 5;           // 8 waves
    const int lane = tid & 31;
    const int kh   = lane >> 4;          // K-half of the A layout
    const int nl   = lane & 15;
    const int i    = ig * 8 + wave;

    // Per-lane epilogue constants: channel h = 16t + nl
    float bias[4], w20[4], w21[4];
    #pragma unroll
    for (int t = 0; t < 4; ++t) {
        int h = 16 * t + nl;
        bias[t] = e1b[h];
        w20[t]  = e2w[h * 2 + 0];
        w21[t]  = e2w[h * 2 + 1];
    }
    const float eb0 = e2b[0], eb1 = e2b[1];

    // Hoist all 16 B fragments into registers (constant for whole kernel)
    v16h bf[4][4];
    #pragma unroll
    for (int s = 0; s < 4; ++s)
        #pragma unroll
        for (int t = 0; t < 4; ++t)
            bf[s][t] = *(const v16h*)&sB[((s * 4 + t) * 32 + lane) * 16];

    // emb_i chunks (constant per wave): chunk m -> dims 16m + 8*kh .. +7
    h8 hiC[4];
    const _Float16* hiPtr = embF16 + ((size_t)b * NN + i) * EMB;
    #pragma unroll
    for (int m = 0; m < 4; ++m)
        hiC[m] = *(const h8*)(hiPtr + 16 * m + 8 * kh);

    float* adjRow = adj + ((size_t)b * NN + i) * NN + jg * 256;
    const int jgbase = jg * 256;

    for (int jp2 = 0; jp2 < 8; ++jp2) {
        const int jb0 = jp2 * 32;        // two 16-j tiles per iteration
        const int jb1 = jb0 + 16;

        h8 hj0[4], hj1[4];
        #pragma unroll
        for (int m = 0; m < 4; ++m) {
            hj0[m] = *(const h8*)&sEmb[(jb0 + nl) * EMB + 16 * m + 8 * kh];
            hj1[m] = *(const h8*)&sEmb[(jb1 + nl) * EMB + 16 * m + 8 * kh];
        }

        v8f accA[4] = {};
        wmma_tile(accA, hiC, hj0, bf);   // issue tile-A WMMAs
        v8f accB[4] = {};
        wmma_tile(accB, hiC, hj1, bf);   // issue tile-B WMMAs

        // tile-A epilogue overlaps tile-B's in-flight WMMAs
        edge_epilogue(accA, bias, w20, w21, eb0, eb1, kh, nl, b, i, jgbase, jb0, adjRow, pred);
        edge_epilogue(accB, bias, w20, w21, eb0, eb1, kh, nl, b, i, jgbase, jb1, adjRow, pred);
    }
}

// ---------------------------------------------------------------------------
extern "C" void kernel_launch(void* const* d_in, const int* in_sizes, int n_in,
                              void* d_out, int out_size, void* d_ws, size_t ws_size,
                              hipStream_t stream) {
    const float* x   = (const float*)d_in[0];
    const float* w1  = (const float*)d_in[1];
    const float* b1  = (const float*)d_in[2];
    const float* w2  = (const float*)d_in[3];
    const float* b2  = (const float*)d_in[4];
    const float* e1w = (const float*)d_in[5];
    const float* e1b = (const float*)d_in[6];
    const float* e2w = (const float*)d_in[7];
    const float* e2b = (const float*)d_in[8];

    float* adj    = (float*)d_out;                                   // BS*N*N
    float* pred   = adj + (size_t)BS * NN * NN;                      // BS*N*(N-1)*2
    float* embF32 = pred + (size_t)BS * NN * (NN - 1) * 2;           // BS*N*EMB
    _Float16* embF16 = (_Float16*)d_ws;                              // 256 KB scratch

    node_mlp_kernel<<<BS * NN, 256, 0, stream>>>(x, w1, b1, w2, b2, embF32, embF16);
    edge_mlp_kernel<<<BS * 64 * 2, 256, 0, stream>>>(embF16, e1w, e1b, e2w, e2b, adj, pred);
}
